// GATLayer_37967510897371
// MI455X (gfx1250) — matrix operand order repaired and checked by reading, not compiled, not run on hardware
//
#include <hip/hip_runtime.h>
#include <hip/hip_bf16.h>
#include <math.h>

#define D 64
#define TWO_D 128

typedef __attribute__((ext_vector_type(16))) __bf16         v16bf;
typedef __attribute__((ext_vector_type(8)))  float          v8f;
typedef __attribute__((ext_vector_type(8)))  unsigned short v8u;
typedef __attribute__((ext_vector_type(16))) unsigned short v16u;

// float -> bf16 bits, round-to-nearest-even
__device__ __forceinline__ unsigned short f2bf(float f) {
  unsigned u = __float_as_uint(f);
  u += 0x7fffu + ((u >> 16) & 1u);
  return (unsigned short)(u >> 16);
}

// CAS-based float atomic max (init value is -inf, so plain float compare is fine)
__device__ __forceinline__ void atomic_max_f32(float* addr, float val) {
  int* ia = (int*)addr;
  int cur = *ia;
  while (__int_as_float(cur) < val) {
    int old = atomicCAS(ia, cur, __float_as_int(val));
    if (old == cur) break;
    cur = old;
  }
}

// ---------------------------------------------------------------------------
// Kernel 1: initialize z = 0, m = -inf, denom = 0 (must run every call;
// the harness does not re-clean buffers between graph replays)
// ---------------------------------------------------------------------------
__global__ void gat_init(float* __restrict__ z, float* __restrict__ m,
                         float* __restrict__ denom, int N, long totalZ) {
  long i = (long)blockIdx.x * blockDim.x + threadIdx.x;
  if (i < totalZ) z[i] = 0.0f;
  if (i < N) {
    m[i] = -__builtin_inff();
    denom[i] = 0.0f;
  }
}

// ---------------------------------------------------------------------------
// Kernel 2: per-edge logits e = w_out . tanh(W_attn @ [h_src;h_dst] + b)
// One 16-edge tile per wave; 16 x v_wmma_f32_16x16x32_bf16 per tile.
// Also folds the segment-max into an atomic max on m[dst].
// ---------------------------------------------------------------------------
__global__ __launch_bounds__(256)
void gat_edge_logits(const float* __restrict__ feat,
                     const float* __restrict__ W_attn,
                     const float* __restrict__ b_attn,
                     const float* __restrict__ w_out,
                     const int*   __restrict__ src,
                     const int*   __restrict__ dst,
                     float* __restrict__ ews,   // [E] edge logits out
                     float* __restrict__ m,     // [N] running segment max
                     int E) {
  // B operand (W_attn^T, 128x64) pre-swizzled to the ISA 32x16 bf16 B layout:
  //   index = (((nt*4 + kc)*32 + lane)*16 + j)
  //   lane 0-15:  n = nt*16+lane,     k = kc*32 + (j&7) + (j&8 ? 16 : 0)
  //   lane 16-31: n = nt*16+lane-16,  k = kc*32 + 8 + (j&7) + (j&8 ? 16 : 0)
  __shared__ __align__(16) unsigned short Bswz[4 * 4 * 32 * 16];  // 16 KB
  // A tiles: per wave, 16 edges x 128 concat features (bf16)
  __shared__ __align__(16) unsigned short Ash[8][16][TWO_D];      // 32 KB

  const int t    = threadIdx.x;
  const int lane = t & 31;
  const int w    = t >> 5;

  // ---- stage B once per block ----
  for (int i = t; i < 4 * 4 * 32 * 16; i += 256) {
    int j  = i & 15;
    int ln = (i >> 4) & 31;
    int kc = (i >> 9) & 3;
    int nt = i >> 11;
    int n  = nt * 16 + (ln & 15);
    int k  = kc * 32 + ((ln < 16) ? 0 : 8) + (j & 7) + ((j & 8) ? 16 : 0);
    Bswz[i] = f2bf(W_attn[n * TWO_D + k]);
  }

  // ---- stage A: gather [h_src | h_dst] for this wave's 16 edges ----
  const int tileBase = (blockIdx.x * 8 + w) * 16;
  for (int i = lane; i < 16 * 32; i += 32) {
    int r = i >> 5;          // edge row 0..15
    int c = (i & 31) * 4;    // concat column 0,4,...,124
    int e = tileBase + r;
    if (e >= E) e = E - 1;
    int node, col;
    if (c < D) { node = src[e]; col = c; }
    else       { node = dst[e]; col = c - D; }
    const float4 v = *(const float4*)(feat + (long)node * D + col);
    uint2 pk;
    pk.x = (unsigned)f2bf(v.x) | ((unsigned)f2bf(v.y) << 16);
    pk.y = (unsigned)f2bf(v.z) | ((unsigned)f2bf(v.w) << 16);
    *(uint2*)&Ash[w][r][c] = pk;
  }
  __syncthreads();

  // ---- 16-edge x 64-out GEMM, K=128, via 4x4 WMMAs ----
  const int arow  = lane & 15;
  const int abase = (lane < 16) ? 0 : 8;  // ISA 16-bit A layout half-wave split

  v8f acc[4] = {};  // 4 N-tiles of the 16x16 f32 accumulator

#pragma unroll
  for (int kc = 0; kc < 4; ++kc) {
    const int kk = kc * 32;
    v8u alo = *(const v8u*)&Ash[w][arow][kk + abase];
    v8u ahi = *(const v8u*)&Ash[w][arow][kk + 16 + abase];
    v16u au = __builtin_shufflevector(alo, ahi, 0, 1, 2, 3, 4, 5, 6, 7,
                                      8, 9, 10, 11, 12, 13, 14, 15);
    v16bf a = __builtin_bit_cast(v16bf, au);
#pragma unroll
    for (int nt = 0; nt < 4; ++nt) {
      const int bidx = (((nt * 4 + kc) * 32) + lane) * 16;
      v8u blo = *(const v8u*)&Bswz[bidx];
      v8u bhi = *(const v8u*)&Bswz[bidx + 8];
      v16u bu = __builtin_shufflevector(blo, bhi, 0, 1, 2, 3, 4, 5, 6, 7,
                                        8, 9, 10, 11, 12, 13, 14, 15);
      v16bf b = __builtin_bit_cast(v16bf, bu);
      acc[nt] = __builtin_amdgcn_wmma_f32_16x16x32_bf16(
          false, a, false, b, (short)0, acc[nt], false, false);
    }
  }

  // ---- epilogue: e[m] = sum_n tanh(acc + b[n]) * w_out[n] ----
  // C/D layout: lanes 0-15 hold (M = vgpr, N = lane); lanes 16-31 hold
  // (M = 8+vgpr, N = lane-16). Reduce over N across each 16-lane half.
  float p[8];
#pragma unroll
  for (int i = 0; i < 8; ++i) p[i] = 0.0f;
#pragma unroll
  for (int nt = 0; nt < 4; ++nt) {
    int   n  = nt * 16 + (lane & 15);
    float bb = b_attn[n];
    float wo = w_out[n];
#pragma unroll
    for (int i = 0; i < 8; ++i) p[i] += tanhf(acc[nt][i] + bb) * wo;
  }
#pragma unroll
  for (int mask = 1; mask <= 8; mask <<= 1) {
#pragma unroll
    for (int i = 0; i < 8; ++i) p[i] += __shfl_xor(p[i], mask, 32);
  }

  if ((lane & 15) == 0) {
    int rb = (lane < 16) ? 0 : 8;
#pragma unroll
    for (int i = 0; i < 8; ++i) {
      int e = tileBase + rb + i;
      if (e < E) {
        ews[e] = p[i];
        atomic_max_f32(&m[dst[e]], p[i]);
      }
    }
  }
}

// ---------------------------------------------------------------------------
// Kernel 3: ex = exp(e - m[dst]); denom[dst] += ex  (native f32 atomic add)
// ---------------------------------------------------------------------------
__global__ void gat_softmax_num(const int* __restrict__ dst,
                                float* __restrict__ ews,
                                float* __restrict__ denom,
                                const float* __restrict__ m, int E) {
  int e = blockIdx.x * blockDim.x + threadIdx.x;
  if (e >= E) return;
  int   d = dst[e];
  float x = expf(ews[e] - m[d]);
  ews[e] = x;  // overwrite logits with numerator
  atomicAdd(&denom[d], x);
}

// ---------------------------------------------------------------------------
// Kernel 4: z[dst] += (ex/denom[dst]) * h_src  — thread per (edge, 4 channels)
// ---------------------------------------------------------------------------
__global__ void gat_aggregate(const float* __restrict__ feat,
                              const int* __restrict__ src,
                              const int* __restrict__ dst,
                              const float* __restrict__ ews,
                              const float* __restrict__ denom,
                              float* __restrict__ z, int E) {
  long t = (long)blockIdx.x * blockDim.x + threadIdx.x;
  long total = (long)E * 16;
  if (t >= total) return;
  int e = (int)(t >> 4);
  int c = (int)(t & 15) * 4;
  int d = dst[e];
  float den   = denom[d];
  float alpha = ews[e] / (den == 0.0f ? 1.0f : den);
  const float4 v = *(const float4*)(feat + (long)src[e] * D + c);
  float* zp = z + (long)d * D + c;
  atomicAdd(zp + 0, alpha * v.x);
  atomicAdd(zp + 1, alpha * v.y);
  atomicAdd(zp + 2, alpha * v.z);
  atomicAdd(zp + 3, alpha * v.w);
}

// ---------------------------------------------------------------------------
extern "C" void kernel_launch(void* const* d_in, const int* in_sizes, int n_in,
                              void* d_out, int out_size, void* d_ws, size_t ws_size,
                              hipStream_t stream) {
  const float* feat  = (const float*)d_in[0];
  const float* Wattn = (const float*)d_in[1];
  const float* battn = (const float*)d_in[2];
  const float* wout  = (const float*)d_in[3];
  const int*   src   = (const int*)d_in[4];
  const int*   dst   = (const int*)d_in[5];

  const int N = in_sizes[0] / D;
  const int E = in_sizes[4];

  float* z     = (float*)d_out;
  float* ews   = (float*)d_ws;        // [E] logits -> numerators (~5.1 MB)
  float* m     = ews + E;             // [N] segment max (~0.4 MB)
  float* denom = m + N;               // [N] segment sum (~0.4 MB)

  const long totalZ = (long)N * D;
  {
    int blocks = (int)((totalZ + 255) / 256);
    gat_init<<<blocks, 256, 0, stream>>>(z, m, denom, N, totalZ);
  }
  {
    int blocks = (E + 127) / 128;  // 8 waves/block, 16 edges/wave
    gat_edge_logits<<<blocks, 256, 0, stream>>>(feat, Wattn, battn, wout,
                                                src, dst, ews, m, E);
  }
  {
    int blocks = (E + 255) / 256;
    gat_softmax_num<<<blocks, 256, 0, stream>>>(dst, ews, denom, m, E);
  }
  {
    long totalA = (long)E * 16;
    int blocks = (int)((totalA + 255) / 256);
    gat_aggregate<<<blocks, 256, 0, stream>>>(feat, src, dst, ews, denom, z, E);
  }
}
